// RDGCNEncoder_v3_6347961663816
// MI455X (gfx1250) — matrix operand order repaired and checked by reading, not compiled, not run on hardware
//
#include <hip/hip_runtime.h>

// ---------------------------------------------------------------------------
// CDNA5 (gfx1250): bf16 WMMA GEMM core (128x128 block, 32x64 wave tile),
// split-K with f32 global atomics, scatter/mean SAGE, elementwise glue.
// ---------------------------------------------------------------------------

typedef __attribute__((ext_vector_type(16))) __bf16 v16bf;
typedef __attribute__((ext_vector_type(8)))  float  v8f;

#define BM 128
#define BN 128
#define BK 32
#define PITCH 80      // LDS row: 32 bf16 (64B) + 16B pad -> conflict-free b128

#define F_ACCUM  1
#define F_ATOMIC 2

union FragBF { uint4 u[2]; v16bf v; };

__device__ __forceinline__ unsigned pk2_bf16(float lo, float hi) {
#if __has_builtin(__builtin_amdgcn_cvt_pk_bf16_f32)
  typedef __attribute__((ext_vector_type(2))) __bf16 v2bf;
  v2bf t = __builtin_amdgcn_cvt_pk_bf16_f32(lo, hi);
  return __builtin_bit_cast(unsigned, t);
#else
  // RNE then pack the two high halves with one v_perm_b32
  unsigned a = __float_as_uint(lo);
  unsigned b = __float_as_uint(hi);
  a = a + 0x7FFFu + ((a >> 16) & 1u);
  b = b + 0x7FFFu + ((b >> 16) & 1u);
  return __builtin_amdgcn_perm(b, a, 0x07060302u);
#endif
}
__device__ __forceinline__ uint4 pk8_bf16(float4 a, float4 b) {
  return make_uint4(pk2_bf16(a.x, a.y), pk2_bf16(a.z, a.w),
                    pk2_bf16(b.x, b.y), pk2_bf16(b.z, b.w));
}

// C[M,N] (+)= alpha * (A[M,K] @ W[N,K]^T + bias[N])
// Grid: (N/128, M/128, kslices). Block: 256 threads (8 wave32, 4x2 wave grid).
__global__ __launch_bounds__(256) void gemm_wmma_bf16(
    const float* __restrict__ A, int lda,
    const float* __restrict__ W, int ldw,
    const float* __restrict__ bias,
    float* __restrict__ C, int ldc,
    int K, const float* __restrict__ alpha_ptr,
    int flags, int steps_per_slice)
{
  __shared__ char lds[2 * (BM + BN) * PITCH];   // 40960 B, double-buffered

  const int tid  = threadIdx.x;
  const int lane = tid & 31;
  const int wave = tid >> 5;
  const int wr   = wave & 3;     // wave row group (32 rows)
  const int wc   = wave >> 2;    // wave col group (64 cols)
  const int half = lane >> 4;
  const int r    = lane & 15;

  const int m0 = blockIdx.y * BM;
  const int n0 = blockIdx.x * BN;

  const int total_steps = K / BK;
  const int s_begin = blockIdx.z * steps_per_slice;
  int s_count = total_steps - s_begin;
  if (s_count > steps_per_slice) s_count = steps_per_slice;
  if (s_count <= 0) return;

  char* sA0 = lds;
  char* sB0 = lds + BM * PITCH;
  char* sA1 = lds + (BM + BN) * PITCH;
  char* sB1 = sA1 + BM * PITCH;

  // staging: both tiles are 128 rows x 32 k; 2 threads per row, 16 floats each
  const int row = tid >> 1, kh = tid & 1;
  const float* aP = A + (size_t)(m0 + row) * lda + kh * 16 + (size_t)s_begin * BK;
  const float* bP = W + (size_t)(n0 + row) * ldw + kh * 16 + (size_t)s_begin * BK;
  const int soff = row * PITCH + kh * 32;

  float4 ra[4], rb[4];

  auto stage = [&](int s) {
    const float4* ap = (const float4*)(aP + (size_t)s * BK);
    const float4* bp = (const float4*)(bP + (size_t)s * BK);
#pragma unroll
    for (int j = 0; j < 4; ++j) { ra[j] = ap[j]; rb[j] = bp[j]; }
  };
  auto commit = [&](char* sa, char* sb) {
    *(uint4*)(sa + soff)      = pk8_bf16(ra[0], ra[1]);
    *(uint4*)(sa + soff + 16) = pk8_bf16(ra[2], ra[3]);
    *(uint4*)(sb + soff)      = pk8_bf16(rb[0], rb[1]);
    *(uint4*)(sb + soff + 16) = pk8_bf16(rb[2], rb[3]);
  };

  v8f acc[2][4];
#pragma unroll
  for (int mg = 0; mg < 2; ++mg)
#pragma unroll
    for (int ng = 0; ng < 4; ++ng) acc[mg][ng] = (v8f){};

  stage(0);
  commit(sA0, sB0);
  __syncthreads();

  for (int s = 0; s < s_count; ++s) {
    const bool more = (s + 1 < s_count);
    if (more) stage(s + 1);
    if (s + 2 < s_count) {
      __builtin_prefetch(aP + (size_t)(s + 2) * BK, 0, 1);   // global_prefetch_b8
      __builtin_prefetch(bP + (size_t)(s + 2) * BK, 0, 1);
    }
    char* sa = (s & 1) ? sA1 : sA0;
    char* sb = (s & 1) ? sB1 : sB0;

    // A fragments 16x32 (ISA 7.12.2): lanes0-15 K=0..7 & 16..23; lanes16-31 K=8..15 & 24..31
    FragBF a[2], b[4];
    {
      const int abase = (wr * 32 + r) * PITCH + half * 16;
      a[0].u[0] = *(const uint4*)(sa + abase);
      a[0].u[1] = *(const uint4*)(sa + abase + 32);
      a[1].u[0] = *(const uint4*)(sa + abase + 16 * PITCH);
      a[1].u[1] = *(const uint4*)(sa + abase + 16 * PITCH + 32);
    }
    // B fragments 32x16: lanes0-15 K=0..7 & 16..23 of col n; lanes16-31 K=8..15 & 24..31
    {
      const int bbase = (wc * 64 + r) * PITCH + half * 16;
#pragma unroll
      for (int ng = 0; ng < 4; ++ng) {
        b[ng].u[0] = *(const uint4*)(sb + bbase + ng * 16 * PITCH);
        b[ng].u[1] = *(const uint4*)(sb + bbase + ng * 16 * PITCH + 32);
      }
    }

#pragma unroll
    for (int mg = 0; mg < 2; ++mg)
#pragma unroll
      for (int ng = 0; ng < 4; ++ng)
        acc[mg][ng] = __builtin_amdgcn_wmma_f32_16x16x32_bf16(
            false, a[mg].v, false, b[ng].v, (short)0, acc[mg][ng], false, false);

    if (more) {
      commit((s & 1) ? sA0 : sA1, (s & 1) ? sB0 : sB1);
      __syncthreads();
    }
  }

  // Epilogue. C/D layout: VGPR v -> row = base + half*8 + v, col = base + r.
  // Pointer-increment form with the uniform mode branch hoisted out of the
  // unrolled store loops (cuts per-store 64-bit address math).
  const float alpha = alpha_ptr ? *alpha_ptr : 1.0f;
  const bool add_bias = (bias != nullptr) && (blockIdx.z == 0);
  float* cbase = C + (size_t)(m0 + wr * 32 + half * 8) * ldc + (n0 + wc * 64 + r);

#pragma unroll
  for (int ng = 0; ng < 4; ++ng) {
    const float bv = add_bias ? bias[n0 + wc * 64 + ng * 16 + r] : 0.0f;
#pragma unroll
    for (int mg = 0; mg < 2; ++mg) {
      float* p = cbase + ng * 16 + (size_t)(mg * 16) * ldc;
      if (flags & F_ATOMIC) {
#pragma unroll
        for (int v = 0; v < 8; ++v) { unsafeAtomicAdd(p, (acc[mg][ng][v] + bv) * alpha); p += ldc; }
      } else if (flags & F_ACCUM) {
#pragma unroll
        for (int v = 0; v < 8; ++v) { *p += (acc[mg][ng][v] + bv) * alpha; p += ldc; }
      } else {
#pragma unroll
        for (int v = 0; v < 8; ++v) { *p = (acc[mg][ng][v] + bv) * alpha; p += ldc; }
      }
    }
  }
}

// ------------------------------ helper kernels ------------------------------

__global__ void zero_kernel(float* __restrict__ p, long long n) {
  long long i = (long long)blockIdx.x * blockDim.x + threadIdx.x;
  long long stride = (long long)gridDim.x * blockDim.x;
  for (; i < n; i += stride) p[i] = 0.0f;
}

__global__ void leaky_kernel(float* __restrict__ p, long long n) {
  long long i = (long long)blockIdx.x * blockDim.x + threadIdx.x;
  long long stride = (long long)gridDim.x * blockDim.x;
  for (; i < n; i += stride) {
    float x = p[i];
    p[i] = x >= 0.0f ? x : 0.01f * x;
  }
}

__global__ void combine2_kernel(const float* __restrict__ x, const float* __restrict__ y,
                                const float* __restrict__ s, int ia, int ib,
                                float* __restrict__ o, long long n) {
  const float a = s[ia], b = s[ib];
  long long i = (long long)blockIdx.x * blockDim.x + threadIdx.x;
  long long stride = (long long)gridDim.x * blockDim.x;
  for (; i < n; i += stride) o[i] = x[i] * a + y[i] * b;
}

__global__ void count_kernel(const int* __restrict__ dst, int E, float* __restrict__ cnt) {
  int e = blockIdx.x * blockDim.x + threadIdx.x;
  if (e < E) unsafeAtomicAdd(&cnt[dst[e]], 1.0f);
}

__global__ void scatter_add_kernel(const float* __restrict__ xsrc,
                                   const int* __restrict__ src, const int* __restrict__ dst,
                                   int D, float* __restrict__ agg) {
  const int e = blockIdx.x;
  const float* xp = xsrc + (size_t)src[e] * D;
  float* ap = agg + (size_t)dst[e] * D;
  for (int j = threadIdx.x; j < D; j += blockDim.x)
    unsafeAtomicAdd(&ap[j], xp[j]);
}

__global__ void meandiv_kernel(float* __restrict__ agg, const float* __restrict__ cnt,
                               int D, long long n) {
  long long i = (long long)blockIdx.x * blockDim.x + threadIdx.x;
  long long stride = (long long)gridDim.x * blockDim.x;
  for (; i < n; i += stride) agg[i] = agg[i] / fmaxf(cnt[i / D], 1.0f);
}

__device__ __forceinline__ void softn(const float* x, int n, float* o) {
  float mx = x[0];
  for (int i = 1; i < n; ++i) mx = fmaxf(mx, x[i]);
  float s = 0.0f;
  for (int i = 0; i < n; ++i) { float e = __expf(x[i] - mx); o[i] = e; s += e; }
  for (int i = 0; i < n; ++i) o[i] /= s;
}

__global__ void softmax6_kernel(const float* wm, const float* wd, const float* wmu,
                                const float* wdu, const float* am, const float* ad,
                                float* __restrict__ o) {
  if (threadIdx.x == 0 && blockIdx.x == 0) {
    softn(wm, 3, o + 0);  softn(wd, 2, o + 3);
    softn(wmu, 3, o + 5); softn(wdu, 2, o + 8);
    softn(am, 2, o + 10); softn(ad, 2, o + 12);
  }
}

// ------------------------------ host orchestration ------------------------------

static inline void k_zero(hipStream_t st, float* p, long long n) {
  int blocks = (int)((n + 255) / 256); if (blocks > 4096) blocks = 4096;
  zero_kernel<<<blocks, 256, 0, st>>>(p, n);
}
static inline void k_leaky(hipStream_t st, float* p, long long n) {
  int blocks = (int)((n + 255) / 256); if (blocks > 4096) blocks = 4096;
  leaky_kernel<<<blocks, 256, 0, st>>>(p, n);
}
static inline void k_gemm(hipStream_t st, const float* A, int lda, const float* W, int ldw,
                          const float* bias, float* C, int ldc, int M, int N, int K,
                          const float* alpha, int flags, int kslices) {
  int total = K / BK;
  int sps = (total + kslices - 1) / kslices;
  if (kslices > 1) flags |= F_ATOMIC;
  dim3 grid(N / BN, M / BM, kslices);
  gemm_wmma_bf16<<<grid, 256, 0, st>>>(A, lda, W, ldw, bias, C, ldc, K, alpha, flags, sps);
}
static inline void k_meandiv(hipStream_t st, float* agg, const float* cnt, int D, long long n) {
  int blocks = (int)((n + 255) / 256); if (blocks > 4096) blocks = 4096;
  meandiv_kernel<<<blocks, 256, 0, st>>>(agg, cnt, D, n);
}

extern "C" void kernel_launch(void* const* d_in, const int* in_sizes, int n_in,
                              void* d_out, int out_size, void* d_ws, size_t ws_size,
                              hipStream_t stream) {
  (void)n_in; (void)out_size; (void)ws_size;
  const int NM = 1024, ND = 512, EMBD = 96000, IND = 256, OUTD = 128;

#define FP(i) ((const float*)d_in[(i)])
  const float* mi_emb = FP(0);
  const float* mi_sim = FP(1);
  const float* mi_ass = FP(2);
  const float* di_sim = FP(3);
  const float* di_ass = FP(4);
  const int* e_md = (const int*)d_in[5];
  const int* e_dm = (const int*)d_in[6];
  const int E = in_sizes[5] / 2;
  const int *md_src = e_md, *md_dst = e_md + E;
  const int *dm_src = e_dm, *dm_dst = e_dm + E;
  // params flattened in dict insertion order: 5 updaters (W,b), 5 raw-lin (W,b),
  // 5 up-lin (W,b), 6 weight vecs, gnn md[0..2] then dm[0..2] each (Wl,bl,Wr)
#define MD_Wl(i) FP(43 + 3*(i))
#define MD_bl(i) FP(44 + 3*(i))
#define MD_Wr(i) FP(45 + 3*(i))
#define DM_Wl(i) FP(52 + 3*(i))
#define DM_bl(i) FP(53 + 3*(i))
#define DM_Wr(i) FP(54 + 3*(i))

  float* ws = (float*)d_ws;
  float* sm     = ws;                         // 16 softmaxed scalars
  float* x_m    = sm + 16;                    // 1024x128
  float* x_d    = x_m + (size_t)NM * OUTD;    // 512x128
  float* ubuf   = x_d + (size_t)ND * OUTD;    // up to 1024x1024
  float* h_m    = ubuf + (size_t)NM * NM;     // up to 1024x512
  float* h_d    = h_m + (size_t)NM * 512;     // up to 512x512
  float* nm     = h_d + (size_t)ND * 512;     // up to 1024x512
  float* nd     = nm + (size_t)NM * 512;      // up to 512x512
  float* mean_m = nd + (size_t)ND * 512;      // up to 1024x512
  float* mean_d = mean_m + (size_t)NM * 512;  // up to 512x512
  float* cnt_m  = mean_d + (size_t)ND * 512;  // 1024
  float* cnt_d  = cnt_m + NM;                 // 512

  // softmax-normalized fusion weights
  softmax6_kernel<<<1, 32, 0, stream>>>(FP(37), FP(38), FP(39), FP(40), FP(41), FP(42), sm);

  // ---- FeatureUpdater branches + updated-feature fusion into h_m / h_d ----
  k_zero(stream, h_m, (long long)NM * IND);
  // upd_m_emb (K=96000, split-K=20 -> 1280 blocks) -> leaky -> m_emb_up
  k_zero(stream, ubuf, (long long)NM * NM);
  k_gemm(stream, mi_emb, EMBD, FP(7), EMBD, FP(8), ubuf, NM, NM, NM, EMBD, nullptr, 0, 20);
  k_leaky(stream, ubuf, (long long)NM * NM);
  k_gemm(stream, ubuf, NM, FP(27), NM, FP(28), h_m, IND, NM, IND, NM, &sm[5], F_ACCUM, 1);
  // upd_m_sim -> m_sim_up
  k_gemm(stream, mi_sim, NM, FP(9), NM, FP(10), ubuf, NM, NM, NM, NM, nullptr, 0, 1);
  k_leaky(stream, ubuf, (long long)NM * NM);
  k_gemm(stream, ubuf, NM, FP(29), NM, FP(30), h_m, IND, NM, IND, NM, &sm[6], F_ACCUM, 1);
  // upd_m_ass -> m_ass_up
  k_gemm(stream, mi_ass, ND, FP(11), ND, FP(12), ubuf, ND, NM, ND, ND, nullptr, 0, 1);
  k_leaky(stream, ubuf, (long long)NM * ND);
  k_gemm(stream, ubuf, ND, FP(31), ND, FP(32), h_m, IND, NM, IND, ND, &sm[7], F_ACCUM, 1);

  k_zero(stream, h_d, (long long)ND * IND);
  // upd_d_sim -> d_sim_up
  k_gemm(stream, di_sim, ND, FP(13), ND, FP(14), ubuf, ND, ND, ND, ND, nullptr, 0, 1);
  k_leaky(stream, ubuf, (long long)ND * ND);
  k_gemm(stream, ubuf, ND, FP(33), ND, FP(34), h_d, IND, ND, IND, ND, &sm[8], F_ACCUM, 1);
  // upd_d_ass -> d_ass_up
  k_gemm(stream, di_ass, NM, FP(15), NM, FP(16), ubuf, NM, ND, NM, NM, nullptr, 0, 1);
  k_leaky(stream, ubuf, (long long)ND * NM);
  k_gemm(stream, ubuf, NM, FP(35), NM, FP(36), h_d, IND, ND, IND, NM, &sm[9], F_ACCUM, 1);

  // ---- raw-feature fusion into x_m / x_d ----
  k_zero(stream, x_m, (long long)NM * OUTD);
  k_gemm(stream, mi_emb, EMBD, FP(17), EMBD, FP(18), x_m, OUTD, NM, OUTD, EMBD, &sm[0], F_ACCUM, 64);
  k_gemm(stream, mi_sim, NM, FP(19), NM, FP(20), x_m, OUTD, NM, OUTD, NM, &sm[1], F_ACCUM, 1);
  k_gemm(stream, mi_ass, ND, FP(21), ND, FP(22), x_m, OUTD, NM, OUTD, ND, &sm[2], F_ACCUM, 1);
  k_zero(stream, x_d, (long long)ND * OUTD);
  k_gemm(stream, di_sim, ND, FP(23), ND, FP(24), x_d, OUTD, ND, OUTD, ND, &sm[3], F_ACCUM, 1);
  k_gemm(stream, di_ass, NM, FP(25), NM, FP(26), x_d, OUTD, ND, OUTD, NM, &sm[4], F_ACCUM, 1);

  // ---- degree counts (same for all 3 layers) ----
  k_zero(stream, cnt_m, NM);
  count_kernel<<<(E + 255) / 256, 256, 0, stream>>>(dm_dst, E, cnt_m);
  k_zero(stream, cnt_d, ND);
  count_kernel<<<(E + 255) / 256, 256, 0, stream>>>(md_dst, E, cnt_d);

  // ---- 3-layer hetero SAGE ----
  const int ind3[3]  = {256, 512, 256};
  const int outd3[3] = {512, 256, 128};
  float *hm = h_m, *hd = h_d, *tm = nm, *td = nd;
  for (int i = 0; i < 3; ++i) {
    const int di = ind3[i], dout = outd3[i];
    // mean over disease->miRNA edges (aggregates hd into miRNA nodes)
    k_zero(stream, mean_m, (long long)NM * di);
    scatter_add_kernel<<<E, 128, 0, stream>>>(hd, dm_src, dm_dst, di, mean_m);
    k_meandiv(stream, mean_m, cnt_m, di, (long long)NM * di);
    // mean over miRNA->disease edges (aggregates hm into disease nodes)
    k_zero(stream, mean_d, (long long)ND * di);
    scatter_add_kernel<<<E, 128, 0, stream>>>(hm, md_src, md_dst, di, mean_d);
    k_meandiv(stream, mean_d, cnt_d, di, (long long)ND * di);
    // nm = mean_m @ Wl_dm^T + bl_dm + hm @ Wr_dm^T
    k_gemm(stream, mean_m, di, DM_Wl(i), di, DM_bl(i), tm, dout, NM, dout, di, nullptr, 0, 1);
    k_gemm(stream, hm,     di, DM_Wr(i), di, nullptr,  tm, dout, NM, dout, di, nullptr, F_ACCUM, 1);
    // nd = mean_d @ Wl_md^T + bl_md + hd @ Wr_md^T
    k_gemm(stream, mean_d, di, MD_Wl(i), di, MD_bl(i), td, dout, ND, dout, di, nullptr, 0, 1);
    k_gemm(stream, hd,     di, MD_Wr(i), di, nullptr,  td, dout, ND, dout, di, nullptr, F_ACCUM, 1);
    if (i < 2) {
      k_leaky(stream, tm, (long long)NM * dout);
      k_leaky(stream, td, (long long)ND * dout);
    }
    float* t;
    t = hm; hm = tm; tm = t;
    t = hd; hd = td; td = t;
  }

  // ---- final combine ----
  float* out_m = (float*)d_out;
  float* out_d = out_m + (size_t)NM * OUTD;
  combine2_kernel<<<512, 256, 0, stream>>>(x_m, hm, sm, 10, 11, out_m, (long long)NM * OUTD);
  combine2_kernel<<<256, 256, 0, stream>>>(x_d, hd, sm, 12, 13, out_d, (long long)ND * OUTD);
}
